// LinformerAttention_72653666779413
// MI455X (gfx1250) — matrix operand order repaired and checked by reading, not compiled, not run on hardware
//
#include <hip/hip_runtime.h>

// ---------- types ----------
typedef __attribute__((ext_vector_type(16))) __bf16 v16bf;
typedef __attribute__((ext_vector_type(8)))  float  v8f;

union FragBF {
    v16bf v;
    uint4 q[2];
};

// Native bf16 conversion: clang folds pairs of fptrunc into packed hw cvt.
__device__ __forceinline__ unsigned short f2bf(float f) {
    union { __bf16 b; unsigned short u; } c;
    c.b = (__bf16)f;
    return c.u;
}

__device__ __forceinline__ unsigned int pack2bf(float lo, float hi) {
    union { __bf16 b[2]; unsigned int u; } c;
    c.b[0] = (__bf16)lo;
    c.b[1] = (__bf16)hi;
    return c.u;
}

// Async global->LDS copy (CDNA5): 16B per lane, tracked by ASYNCcnt.
__device__ __forceinline__ void async_copy_b128(unsigned lds_byte_off, const void* gaddr) {
    asm volatile("global_load_async_to_lds_b128 %0, %1, off"
                 :: "v"(lds_byte_off), "v"(gaddr) : "memory");
}
__device__ __forceinline__ void async_wait0() {
    asm volatile("s_wait_asynccnt 0x0" ::: "memory");
}

#define WMMA_BF16(A, B, C) \
    __builtin_amdgcn_wmma_f32_16x16x32_bf16(false, (A), false, (B), (short)0, (C), false, false)

// =====================================================================
// Kernel 1/4: C[M,1024] = A[M,1024] @ W^T + bias
//   OUT_MODE 0: bf16, layout [b,h,n,d]   (Q)
//   OUT_MODE 1: bf16, layout [b,h,d,n]   (K^T / V^T)
//   OUT_MODE 2: f32,  layout row-major   (final output)
// Block tile 128x128, 8 waves (4x2), wave tile 32x64 (2x2x4 WMMA tiles)
// =====================================================================
template<int OUT_MODE, bool A_IS_BF16>
__global__ __launch_bounds__(256)
void gemm_proj(const void* __restrict__ Aptr, const float* __restrict__ W,
               const float* __restrict__ bias, void* __restrict__ Out) {
    constexpr int KDIM = 1024;
    constexpr int LDA  = 40; // bf16 elems per LDS row (32 + pad), 80B, 16B-aligned

    __shared__ __attribute__((aligned(16))) unsigned short sA[128 * LDA];
    __shared__ __attribute__((aligned(16))) unsigned short sB[128 * LDA];

    const int tid  = threadIdx.x;
    const int wave = tid >> 5;
    const int lane = tid & 31;
    const int blockM = blockIdx.x * 128;
    const int blockN = blockIdx.y * 128;
    const int waveM = (wave & 3) * 32;
    const int waveN = (wave >> 2) * 64;
    const int lh = lane & 15;        // row/col within 16x16 tile
    const int hh = lane >> 4;        // lane half

    v8f acc[8] = {};

    for (int kt = 0; kt < KDIM; kt += 32) {
        // ---- stage A tile 128x32 -> bf16 LDS ----
        if (A_IS_BF16) {
            // already bf16: async DMA global -> LDS, 16B chunks (4 per row)
            const unsigned short* Ab = (const unsigned short*)Aptr;
            #pragma unroll
            for (int i = 0; i < 2; ++i) {
                int idx = tid + i * 256;          // 0..511
                int row = idx >> 2, c = idx & 3;  // 4 x 16B per 64B row-chunk
                const void* g = Ab + (size_t)(blockM + row) * KDIM + kt + c * 8;
                unsigned lds = (unsigned)(uintptr_t)&sA[row * LDA + c * 8];
                async_copy_b128(lds, g);
            }
        } else {
            const float* Af = (const float*)Aptr;
            #pragma unroll
            for (int i = 0; i < 8; ++i) {
                int idx = tid + i * 256;
                int row = idx >> 4, c2 = idx & 15;
                const float* p = Af + (size_t)(blockM + row) * KDIM + kt + c2 * 2;
                *(unsigned int*)&sA[row * LDA + c2 * 2] = pack2bf(p[0], p[1]);
            }
        }
        // ---- stage B tile: rows = output cols, B[n][k] = W[n][k] ----
        #pragma unroll
        for (int i = 0; i < 8; ++i) {
            int idx = tid + i * 256;
            int row = idx >> 4, c2 = idx & 15;
            const float* p = W + (size_t)(blockN + row) * KDIM + kt + c2 * 2;
            *(unsigned int*)&sB[row * LDA + c2 * 2] = pack2bf(p[0], p[1]);
        }
        if (A_IS_BF16) async_wait0();
        __syncthreads();

        // ---- fragments ----
        FragBF a[2], b[4];
        #pragma unroll
        for (int tm = 0; tm < 2; ++tm) {
            int row = waveM + tm * 16 + lh;
            int k0  = hh << 3;
            a[tm].q[0] = *(const uint4*)&sA[row * LDA + k0];
            a[tm].q[1] = *(const uint4*)&sA[row * LDA + k0 + 16];
        }
        #pragma unroll
        for (int tn = 0; tn < 4; ++tn) {
            int row = waveN + tn * 16 + lh;
            int k0  = hh << 3;
            b[tn].q[0] = *(const uint4*)&sB[row * LDA + k0];
            b[tn].q[1] = *(const uint4*)&sB[row * LDA + k0 + 16];
        }
        #pragma unroll
        for (int tm = 0; tm < 2; ++tm)
            #pragma unroll
            for (int tn = 0; tn < 4; ++tn)
                acc[tm * 4 + tn] = WMMA_BF16(a[tm].v, b[tn].v, acc[tm * 4 + tn]);
        __syncthreads();
    }

    // ---- epilogue: bias + scatter ----
    #pragma unroll
    for (int tm = 0; tm < 2; ++tm) {
        #pragma unroll
        for (int tn = 0; tn < 4; ++tn) {
            int gc = blockN + waveN + tn * 16 + lh;
            float bv = bias[gc];
            #pragma unroll
            for (int j = 0; j < 8; ++j) {
                int gr = blockM + waveM + tm * 16 + (hh << 3) + j;
                float val = acc[tm * 4 + tn][j] + bv;
                if (OUT_MODE == 2) {
                    ((float*)Out)[(size_t)gr * 1024 + gc] = val;
                } else {
                    int b = gr >> 12, n = gr & 4095;
                    int h = gc >> 6,  d = gc & 63;
                    if (OUT_MODE == 0)
                        ((unsigned short*)Out)[(((size_t)(b * 16 + h) * 4096 + n) << 6) + d] = f2bf(val);
                    else
                        ((unsigned short*)Out)[(((size_t)(b * 16 + h) * 64 + d) << 12) + n] = f2bf(val);
                }
            }
        }
    }
}

// =====================================================================
// Kernel 2: per (b,h): P[256,64] = E[h](256x4096) @ X^T(4096x64)
//   X^T stored bf16 [bh][64][4096]. TRANSPOSED=0 -> out [bh][k][d] (Kp)
//   TRANSPOSED=1 -> out [bh][d][k] (Vp^T). Block: 128 k-rows x 64 d.
// =====================================================================
template<int TRANSPOSED>
__global__ __launch_bounds__(256)
void kv_project(const float* __restrict__ E, const unsigned short* __restrict__ Xt,
                unsigned short* __restrict__ Outp) {
    constexpr int LDA = 40;
    __shared__ __attribute__((aligned(16))) unsigned short sA[128 * LDA];
    __shared__ __attribute__((aligned(16))) unsigned short sB[64 * LDA];

    const int tid = threadIdx.x, wave = tid >> 5, lane = tid & 31;
    const int bh = blockIdx.x, h = bh & 15;
    const int blockM = blockIdx.y * 128;
    const int waveM = (wave & 3) * 32;
    const int waveN = (wave >> 2) * 32;
    const int lh = lane & 15, hh = lane >> 4;

    const float* Eh = E + (size_t)h * 256 * 4096;
    const unsigned short* Xb = Xt + (size_t)bh * 64 * 4096;

    v8f acc[4] = {};

    for (int kt = 0; kt < 4096; kt += 32) {
        // ---- A tile (E, fp32 -> bf16) ----
        #pragma unroll
        for (int i = 0; i < 8; ++i) {
            int idx = tid + i * 256;
            int row = idx >> 4, c2 = idx & 15;
            const float* p = Eh + (size_t)(blockM + row) * 4096 + kt + c2 * 2;
            *(unsigned int*)&sA[row * LDA + c2 * 2] = pack2bf(p[0], p[1]);
        }
        // ---- B tile (bf16): async DMA, 64 rows x 64B = 256 x 16B chunks ----
        {
            int row = tid >> 2, c = tid & 3;
            const void* g = Xb + (size_t)row * 4096 + kt + c * 8;
            unsigned lds = (unsigned)(uintptr_t)&sB[row * LDA + c * 8];
            async_copy_b128(lds, g);
        }
        async_wait0();
        __syncthreads();

        FragBF a[2], b[2];
        #pragma unroll
        for (int tm = 0; tm < 2; ++tm) {
            int row = waveM + tm * 16 + lh, k0 = hh << 3;
            a[tm].q[0] = *(const uint4*)&sA[row * LDA + k0];
            a[tm].q[1] = *(const uint4*)&sA[row * LDA + k0 + 16];
        }
        #pragma unroll
        for (int tn = 0; tn < 2; ++tn) {
            int row = waveN + tn * 16 + lh, k0 = hh << 3;
            b[tn].q[0] = *(const uint4*)&sB[row * LDA + k0];
            b[tn].q[1] = *(const uint4*)&sB[row * LDA + k0 + 16];
        }
        #pragma unroll
        for (int tm = 0; tm < 2; ++tm)
            #pragma unroll
            for (int tn = 0; tn < 2; ++tn)
                acc[tm * 2 + tn] = WMMA_BF16(a[tm].v, b[tn].v, acc[tm * 2 + tn]);
        __syncthreads();
    }

    unsigned short* Ob = Outp + (size_t)bh * 16384;
    #pragma unroll
    for (int tm = 0; tm < 2; ++tm)
        #pragma unroll
        for (int tn = 0; tn < 2; ++tn)
            #pragma unroll
            for (int j = 0; j < 8; ++j) {
                int gr = blockM + waveM + tm * 16 + (hh << 3) + j; // k index
                int gc = waveN + tn * 16 + lh;                     // d index
                unsigned short bv = f2bf(acc[tm * 2 + tn][j]);
                if (TRANSPOSED) Ob[(size_t)gc * 256 + gr] = bv;
                else            Ob[(size_t)gr * 64 + gc]  = bv;
            }
}

// =====================================================================
// Kernel 3: attention per (bh, 64-row tile). 4 waves, 16 Q-rows each.
//  S = Q.Kp^T -> softmax (in-register, shfl reductions) -> C = P.Vp
// =====================================================================
__global__ __launch_bounds__(128)
void attention(const unsigned short* __restrict__ Qb, const unsigned short* __restrict__ Kp,
               const unsigned short* __restrict__ VpT, unsigned short* __restrict__ Ctx) {
    constexpr int LDP = 264; // 256 + pad, multiple of 8
    __shared__ __attribute__((aligned(16))) unsigned short sP[64 * LDP];

    const int bh = blockIdx.x;
    const int nblk = blockIdx.y * 64;
    const int wave = threadIdx.x >> 5, lane = threadIdx.x & 31;
    const int lh = lane & 15, hh = lane >> 4;
    const int wrow = wave * 16;

    const unsigned short* Qbase = Qb  + (size_t)bh * 4096 * 64;
    const unsigned short* Kbase = Kp  + (size_t)bh * 16384;
    const unsigned short* Vbase = VpT + (size_t)bh * 16384;

    // ---- phase 1: scores S[16 tiles of 16x16] over k=256, d-loop 2x32 ----
    v8f S[16] = {};
    #pragma unroll
    for (int s = 0; s < 2; ++s) {
        FragBF a;
        int row = nblk + wrow + lh;
        int k0  = s * 32 + (hh << 3);
        a.q[0] = *(const uint4*)(Qbase + (size_t)row * 64 + k0);
        a.q[1] = *(const uint4*)(Qbase + (size_t)row * 64 + k0 + 16);
        #pragma unroll
        for (int t = 0; t < 16; ++t) {
            FragBF b;
            int kk = t * 16 + lh;            // k-proj index (B column)
            int d0 = s * 32 + (hh << 3);
            b.q[0] = *(const uint4*)(Kbase + (size_t)kk * 64 + d0);
            b.q[1] = *(const uint4*)(Kbase + (size_t)kk * 64 + d0 + 16);
            S[t] = WMMA_BF16(a.v, b.v, S[t]);
        }
    }

    // ---- phase 2: softmax over k (rows live at VGPR j, lane-half hh) ----
    const float cexp = 0.125f * 1.44269504088896f; // (1/sqrt(64)) * log2(e)
    float rs[8];
    #pragma unroll
    for (int j = 0; j < 8; ++j) {
        float m = S[0][j];
        #pragma unroll
        for (int t = 1; t < 16; ++t) m = fmaxf(m, S[t][j]);
        #pragma unroll
        for (int mask = 1; mask < 16; mask <<= 1) m = fmaxf(m, __shfl_xor(m, mask, 32));
        float sum = 0.f;
        #pragma unroll
        for (int t = 0; t < 16; ++t) {
            float p = exp2f((S[t][j] - m) * cexp);
            S[t][j] = p;
            sum += p;
        }
        #pragma unroll
        for (int mask = 1; mask < 16; mask <<= 1) sum += __shfl_xor(sum, mask, 32);
        rs[j] = 1.0f / sum;
        int rowL = wrow + (hh << 3) + j;
        #pragma unroll
        for (int t = 0; t < 16; ++t)
            sP[rowL * LDP + t * 16 + lh] = f2bf(S[t][j]);
    }
    __syncthreads();

    // ---- phase 3: C = P(16x256) @ Vp(256x64), k-loop 8x32 ----
    v8f C4[4] = {};
    #pragma unroll
    for (int st = 0; st < 8; ++st) {
        FragBF a;
        int row = wrow + lh;
        int k0  = st * 32 + (hh << 3);
        a.q[0] = *(const uint4*)&sP[row * LDP + k0];
        a.q[1] = *(const uint4*)&sP[row * LDP + k0 + 16];
        #pragma unroll
        for (int tn = 0; tn < 4; ++tn) {
            FragBF b;
            int d   = tn * 16 + lh;
            int kk0 = st * 32 + (hh << 3);
            b.q[0] = *(const uint4*)(Vbase + (size_t)d * 256 + kk0);
            b.q[1] = *(const uint4*)(Vbase + (size_t)d * 256 + kk0 + 16);
            C4[tn] = WMMA_BF16(a.v, b.v, C4[tn]);
        }
    }

    // ---- write ctx bf16 [b][n][h*64+d] ----
    const int b = bh >> 4, h = bh & 15;
    #pragma unroll
    for (int tn = 0; tn < 4; ++tn)
        #pragma unroll
        for (int j = 0; j < 8; ++j) {
            int n = nblk + wrow + (hh << 3) + j;
            int d = tn * 16 + lh;
            float val = C4[tn][j] * rs[j];
            Ctx[(((size_t)b * 4096 + n) << 10) + h * 64 + d] = f2bf(val);
        }
}

// =====================================================================
extern "C" void kernel_launch(void* const* d_in, const int* in_sizes, int n_in,
                              void* d_out, int out_size, void* d_ws, size_t ws_size,
                              hipStream_t stream) {
    const float* x  = (const float*)d_in[0];
    const float* Wq = (const float*)d_in[1];
    const float* bq = (const float*)d_in[2];
    const float* Wk = (const float*)d_in[3];
    const float* bk = (const float*)d_in[4];
    const float* Wv = (const float*)d_in[5];
    const float* bv = (const float*)d_in[6];
    const float* Wo = (const float*)d_in[7];
    const float* bo = (const float*)d_in[8];
    const float* E  = (const float*)d_in[9];
    float* out = (float*)d_out;

    char* ws = (char*)d_ws;
    const size_t SZ = 33554432; // 16384*1024*2 bytes (bf16)
    unsigned short* Qb  = (unsigned short*)(ws);
    unsigned short* Kt  = (unsigned short*)(ws + SZ);
    unsigned short* Vt  = (unsigned short*)(ws + 2 * SZ);
    unsigned short* Kpp = (unsigned short*)(ws + 3 * SZ);
    unsigned short* VpT = (unsigned short*)(ws + 3 * SZ + 2097152);
    unsigned short* Ctx = Kt; // Kt is dead after kv_project; reuse

    dim3 b256(256);
    dim3 g1(128, 8); // M=16384/128, N=1024/128
    gemm_proj<0, false><<<g1, b256, 0, stream>>>(x, Wq, bq, Qb);
    gemm_proj<1, false><<<g1, b256, 0, stream>>>(x, Wk, bk, Kt);
    gemm_proj<1, false><<<g1, b256, 0, stream>>>(x, Wv, bv, Vt);

    dim3 g2(64, 2); // 64 (b,h) x (256/128)
    kv_project<0><<<g2, b256, 0, stream>>>(E, Kt, Kpp);
    kv_project<1><<<g2, b256, 0, stream>>>(E, Vt, VpT);

    dim3 g3(64, 64); // 64 (b,h) x (4096/64)
    attention<<<g3, dim3(128), 0, stream>>>(Qb, Kpp, VpT, Ctx);

    dim3 g4(128, 8);
    gemm_proj<2, true><<<g4, b256, 0, stream>>>(Ctx, Wo, bo, out);
}
